// CGNNet_13881334300807
// MI455X (gfx1250) — matrix operand order repaired and checked by reading, not compiled, not run on hardware
//
#include <hip/hip_runtime.h>
#include <cstddef>
#include <cstdint>

// Problem constants (match reference)
#define N_NODES 100000
#define N_EDGES 1600000
#define CH      20           // channels
#define ED      8            // edge feature dim
#define NG      1024         // graphs
#define KREAL   48           // 2*CH + ED
#define EPSV    1e-5f

#define LOG2E_F 1.4426950408889634f
#define LN2_F   0.6931471805599453f

typedef __attribute__((ext_vector_type(16))) _Float16 v16h;
typedef __attribute__((ext_vector_type(8)))  _Float16 v8h;
typedef __attribute__((ext_vector_type(8)))  float    v8f;

// Raw hardware transcendentals: v_exp_f32 (exp2), v_log_f32 (log2), v_rcp_f32.
__device__ __forceinline__ float hexp(float x)  { return __builtin_amdgcn_exp2f(x * LOG2E_F); }
__device__ __forceinline__ float hrcp(float x)  { return __builtin_amdgcn_rcpf(x); }

__device__ __forceinline__ float sigmoid_f(float x) {
    return hrcp(1.0f + hexp(-x));                        // exp2 + add + rcp
}
// softplus(x) = max(x,0) + ln(1 + exp(-|x|)); branchless + stable; arg >= 1 so raw log ok
__device__ __forceinline__ float softplus_f(float x) {
    return fmaxf(x, 0.0f) + LN2_F * __builtin_amdgcn_logf(1.0f + hexp(-fabsf(x)));
}
// tanh(x) = sign(x) * (1 - 2/(exp(2|x|)+1)); stable for large |x| (exp->inf => 1)
__device__ __forceinline__ float tanh_f(float x) {
    float t = 1.0f - 2.0f * hrcp(hexp(2.0f * fabsf(x)) + 1.0f);
    return copysignf(t, x);
}

#define WPB 4   // waves per block in the message kernel (wave32 -> 128 threads)

// ---------------------------------------------------------------------------
// CGConv message + scatter:  per 16-edge tile, z[16x48] @ W[48x20] twice
// (gate + softplus branches) via v_wmma_f32_16x16x32_f16, then
// sigmoid*softplus and global f32 atomic scatter-add into agg[dst].
// ---------------------------------------------------------------------------
__global__ __launch_bounds__(WPB * 32) void kmsg(
    const float* __restrict__ x, const int* __restrict__ ei,
    const float* __restrict__ ea,
    const float* __restrict__ lfw, const float* __restrict__ lfb,
    const float* __restrict__ lsw, const float* __restrict__ lsb,
    float* __restrict__ agg)
{
    __shared__ __align__(16) _Float16 zbuf[WPB][16][64];   // 16 edges x K(pad 64)
    __shared__ int dbuf[WPB][16];                          // dst node per edge row

    const int lane = threadIdx.x & 31;
    const int wave = threadIdx.x >> 5;
    const int nlo  = lane & 15;   // N within n-tile / edge row selector
    const int hsel = lane >> 4;   // half-wave selector

    // ---- Pack B fragments (weights) once: [kt][nt], K = kt*32 + hsel*8 + h (h<8), kt*32 + 16 + hsel*8 + (h-8) (h>=8)
    v16h bF[2][2], bS[2][2];
    for (int kt = 0; kt < 2; ++kt)
        for (int nt = 0; nt < 2; ++nt)
            for (int h = 0; h < 16; ++h) {
                int k = kt * 32 + hsel * 8 + (h < 8 ? h : 16 + (h - 8));
                int n = nt * 16 + nlo;
                float wf = (k < KREAL && n < CH) ? lfw[k * CH + n] : 0.0f;
                float ws = (k < KREAL && n < CH) ? lsw[k * CH + n] : 0.0f;
                bF[kt][nt][h] = (_Float16)wf;
                bS[kt][nt][h] = (_Float16)ws;
            }
    // Biases per output column
    const float bf0 = lfb[nlo];
    const float bs0 = lsb[nlo];
    const float bf1 = (16 + nlo < CH) ? lfb[16 + nlo] : 0.0f;
    const float bs1 = (16 + nlo < CH) ? lsb[16 + nlo] : 0.0f;

    const int ntiles = (N_EDGES + 15) / 16;
    const int stride = gridDim.x * WPB;
    const int iters  = (ntiles + stride - 1) / stride;   // uniform across block
    int tile = blockIdx.x * WPB + wave;

    for (int it = 0; it < iters; ++it, tile += stride) {
        const bool active = (tile < ntiles);            // wave-uniform
        if (active) {
            // ---- Stage z tile into per-wave LDS region (2 lanes per edge row)
            const int m = nlo;
            const int e = tile * 16 + m;
            const bool eok = (e < N_EDGES);
            const int sn = eok ? ei[e] : 0;             // source (x_j)
            const int dn = eok ? ei[N_EDGES + e] : 0;   // target (x_i)
            _Float16* row = &zbuf[wave][m][0];
            if (hsel == 0) {
                dbuf[wave][m] = eok ? dn : -1;
                for (int c = 0; c < CH; ++c)
                    row[c] = (_Float16)(eok ? x[dn * CH + c] : 0.0f);
                for (int c = 0; c < 12; ++c)
                    row[CH + c] = (_Float16)(eok ? x[sn * CH + c] : 0.0f);
            } else {
                for (int c = 12; c < CH; ++c)
                    row[CH + c] = (_Float16)(eok ? x[sn * CH + c] : 0.0f);
                for (int d = 0; d < ED; ++d)
                    row[2 * CH + d] = (_Float16)(eok ? ea[e * ED + d] : 0.0f);
                for (int k = KREAL; k < 64; ++k)
                    row[k] = (_Float16)0.0f;
            }
        }
        __syncthreads();   // order LDS staging vs fragment reads (all threads reach)
        if (active) {
            // ---- A fragments per ISA 16-bit A layout (two ds_load_b128 each)
            const _Float16* zr = &zbuf[wave][nlo][0];
            const int kb = hsel * 8;
            v8h p0 = *(const v8h*)(zr + kb);
            v8h p1 = *(const v8h*)(zr + kb + 16);
            v8h p2 = *(const v8h*)(zr + kb + 32);
            v8h p3 = *(const v8h*)(zr + kb + 48);
            v16h a0, a1;
            for (int i = 0; i < 8; ++i) {
                a0[i] = p0[i]; a0[i + 8] = p1[i];
                a1[i] = p2[i]; a1[i + 8] = p3[i];
            }
            // ---- 8 WMMAs: {gate, softplus} x {ntile0, ntile1}, K chained 0..63
            v8f f0 = {}, f1 = {}, s0 = {}, s1 = {};
            f0 = __builtin_amdgcn_wmma_f32_16x16x32_f16(false, a0, false, bF[0][0], (short)0, f0, false, false);
            f0 = __builtin_amdgcn_wmma_f32_16x16x32_f16(false, a1, false, bF[1][0], (short)0, f0, false, false);
            f1 = __builtin_amdgcn_wmma_f32_16x16x32_f16(false, a0, false, bF[0][1], (short)0, f1, false, false);
            f1 = __builtin_amdgcn_wmma_f32_16x16x32_f16(false, a1, false, bF[1][1], (short)0, f1, false, false);
            s0 = __builtin_amdgcn_wmma_f32_16x16x32_f16(false, a0, false, bS[0][0], (short)0, s0, false, false);
            s0 = __builtin_amdgcn_wmma_f32_16x16x32_f16(false, a1, false, bS[1][0], (short)0, s0, false, false);
            s1 = __builtin_amdgcn_wmma_f32_16x16x32_f16(false, a0, false, bS[0][1], (short)0, s1, false, false);
            s1 = __builtin_amdgcn_wmma_f32_16x16x32_f16(false, a1, false, bS[1][1], (short)0, s1, false, false);
            // ---- Epilogue: activation + scatter-add (C/D row = r + 8*hsel)
            // Fully unrolled so accumulator indexing is static (no v_movrels loop).
#pragma unroll
            for (int r = 0; r < 8; ++r) {
                const int m  = r + 8 * hsel;
                const int dn = dbuf[wave][m];
                if (dn >= 0) {
                    float msg0 = sigmoid_f(f0[r] + bf0) * softplus_f(s0[r] + bs0);
                    atomicAdd(&agg[dn * CH + nlo], msg0);
                    if (nlo < CH - 16) {
                        float msg1 = sigmoid_f(f1[r] + bf1) * softplus_f(s1[r] + bs1);
                        atomicAdd(&agg[dn * CH + 16 + nlo], msg1);
                    }
                }
            }
        }
    }
}

// ---------------------------------------------------------------------------
// agg = x (residual init); block 0 also zeroes the BN stats accumulator.
// ---------------------------------------------------------------------------
__global__ void kcopy(const float* __restrict__ x, float* __restrict__ agg,
                      float* __restrict__ stats)
{
    if (blockIdx.x == 0 && threadIdx.x < 2 * CH) stats[threadIdx.x] = 0.0f;
    const int total = N_NODES * CH;
    for (int i = blockIdx.x * blockDim.x + threadIdx.x; i < total;
         i += gridDim.x * blockDim.x)
        agg[i] = x[i];
}

// ---------------------------------------------------------------------------
// Per-channel sum / sum-of-squares (LDS-binned, one global atomic per channel
// per block) for training-mode BatchNorm stats.
// ---------------------------------------------------------------------------
__global__ void kstats(const float* __restrict__ agg, float* __restrict__ stats)
{
    __shared__ float ssum[CH];
    __shared__ float ssq[CH];
    if (threadIdx.x < CH) { ssum[threadIdx.x] = 0.0f; ssq[threadIdx.x] = 0.0f; }
    __syncthreads();
    const int total = N_NODES * CH;
    for (int i = blockIdx.x * blockDim.x + threadIdx.x; i < total;
         i += gridDim.x * blockDim.x) {
        float v = agg[i];
        int c = i % CH;
        atomicAdd(&ssum[c], v);
        atomicAdd(&ssq[c], v * v);
    }
    __syncthreads();
    if (threadIdx.x < CH) {
        atomicAdd(&stats[threadIdx.x], ssum[threadIdx.x]);
        atomicAdd(&stats[CH + threadIdx.x], ssq[threadIdx.x]);
    }
}

// ---------------------------------------------------------------------------
// x = tanh((agg - mean) * rsqrt(var + eps) * g + b), biased variance.
// ---------------------------------------------------------------------------
__global__ void knorm(const float* __restrict__ agg, const float* __restrict__ stats,
                      const float* __restrict__ g, const float* __restrict__ b,
                      float* __restrict__ xout)
{
    const float invn = 1.0f / (float)N_NODES;
    const int total = N_NODES * CH;
    for (int i = blockIdx.x * blockDim.x + threadIdx.x; i < total;
         i += gridDim.x * blockDim.x) {
        int c = i % CH;
        float mean = stats[c] * invn;
        float var  = stats[CH + c] * invn - mean * mean;
        xout[i] = tanh_f((agg[i] - mean) * rsqrtf(var + EPSV) * g[c] + b[c]);
    }
}

__global__ void kzero(float* __restrict__ p, int n)
{
    int i = blockIdx.x * blockDim.x + threadIdx.x;
    if (i < n) p[i] = 0.0f;
}

// global_add_pool: pooled[batch[n]] += x[n]
__global__ void kpool(const float* __restrict__ x, const int* __restrict__ batch,
                      float* __restrict__ pooled)
{
    const int total = N_NODES * CH;
    for (int i = blockIdx.x * blockDim.x + threadIdx.x; i < total;
         i += gridDim.x * blockDim.x) {
        int n = i / CH, c = i % CH;
        atomicAdd(&pooled[batch[n] * CH + c], x[i]);
    }
}

// MLP head 20 -> 32 -> 8 -> 1 (one graph per thread; weights are tiny/uniform)
__global__ void kmlp(const float* __restrict__ pooled,
                     const float* __restrict__ w1, const float* __restrict__ b1,
                     const float* __restrict__ w2, const float* __restrict__ b2,
                     const float* __restrict__ w3, const float* __restrict__ b3,
                     float* __restrict__ out)
{
    int g = blockIdx.x * blockDim.x + threadIdx.x;
    if (g >= NG) return;
    float p[CH];
    for (int c = 0; c < CH; ++c) p[c] = pooled[g * CH + c];
    float h2[8];
    for (int j = 0; j < 8; ++j) h2[j] = b2[j];
    for (int i = 0; i < 32; ++i) {
        float t = b1[i];
        for (int k = 0; k < CH; ++k) t += p[k] * w1[k * 32 + i];
        t = tanh_f(t);
        for (int j = 0; j < 8; ++j) h2[j] += t * w2[i * 8 + j];
    }
    float o = b3[0];
    for (int j = 0; j < 8; ++j) o += tanh_f(h2[j]) * w3[j];
    out[g] = o;
}

// ---------------------------------------------------------------------------
extern "C" void kernel_launch(void* const* d_in, const int* in_sizes, int n_in,
                              void* d_out, int out_size, void* d_ws, size_t ws_size,
                              hipStream_t stream)
{
    const float* x_in  = (const float*)d_in[0];
    const int*   ei    = (const int*)d_in[1];
    const float* ea    = (const float*)d_in[2];
    const int*   batch = (const int*)d_in[3];
    const float* lfw   = (const float*)d_in[4];
    const float* lfb   = (const float*)d_in[5];
    const float* lsw   = (const float*)d_in[6];
    const float* lsb   = (const float*)d_in[7];
    const float* bng   = (const float*)d_in[8];
    const float* bnb   = (const float*)d_in[9];
    const float* w1    = (const float*)d_in[10];
    const float* b1    = (const float*)d_in[11];
    const float* w2    = (const float*)d_in[12];
    const float* b2    = (const float*)d_in[13];
    const float* w3    = (const float*)d_in[14];
    const float* b3    = (const float*)d_in[15];
    float* out = (float*)d_out;

    // Workspace layout: agg[N*C] | xbuf[N*C] | stats[2*C] | pooled[G*C]
    float* agg    = (float*)d_ws;
    float* xbuf   = agg  + (size_t)N_NODES * CH;
    float* stats  = xbuf + (size_t)N_NODES * CH;
    float* pooled = stats + 2 * CH;

    const int WMAT = KREAL * CH;   // 960 weights per (layer, matrix)
    for (int l = 0; l < 2; ++l) {
        const float* xin = (l == 0) ? x_in : xbuf;
        kcopy <<<1024, 256, 0, stream>>>(xin, agg, stats);
        kmsg  <<<2048, WPB * 32, 0, stream>>>(xin, ei, ea,
                lfw + l * WMAT, lfb + l * CH, lsw + l * WMAT, lsb + l * CH, agg);
        kstats<<<1024, 256, 0, stream>>>(agg, stats);
        knorm <<<1024, 256, 0, stream>>>(agg, stats, bng + l * CH, bnb + l * CH, xbuf);
    }
    kzero<<<(NG * CH + 255) / 256, 256, 0, stream>>>(pooled, NG * CH);
    kpool<<<1024, 256, 0, stream>>>(xbuf, batch, pooled);
    kmlp <<<(NG + 255) / 256, 256, 0, stream>>>(pooled, w1, b1, w2, b2, w3, b3, out);
}
